// TranscendentAttention_60670708023698
// MI455X (gfx1250) — matrix operand order repaired and checked by reading, not compiled, not run on hardware
//
#include <hip/hip_runtime.h>
#include <hip/hip_bf16.h>

typedef __bf16 bf16;
typedef __attribute__((ext_vector_type(16))) __bf16 v16bf;
typedef __attribute__((ext_vector_type(8)))  __bf16 v8bf;
typedef __attribute__((ext_vector_type(8)))  float   v8f;
typedef __attribute__((ext_vector_type(4)))  unsigned v4u;
typedef __attribute__((ext_vector_type(8)))  unsigned v8u;

#define WMMA_BF16(A, B, C) \
  __builtin_amdgcn_wmma_f32_16x16x32_bf16(false, (A), false, (B), (short)0, (C), false, false)

static constexpr int S  = 4096;
static constexpr int D  = 2048;
static constexpr int H  = 16;
static constexpr int HD = 128;
static constexpr int WIN = 2048;

// ---------------------------------------------------------------------------
// WMMA fragment loaders (ISA 7.12.2, wave32, 16-bit operands)
// A 16x32 (MxK): lane<16  -> row=lane,    K = {0..7, 16..23}
//                lane>=16 -> row=lane-16, K = {8..15, 24..31}
// ---------------------------------------------------------------------------
__device__ __forceinline__ v16bf load_A_frag(const bf16* __restrict__ base, int ld, int lane) {
  const int half = lane >> 4;
  const int r    = lane & 15;
  const bf16* p  = base + (size_t)r * ld + half * 8;
  v8bf lo = *(const v8bf*)(p);        // K = half*8 .. +7
  v8bf hi = *(const v8bf*)(p + 16);   // K = half*8+16 .. +23
  return __builtin_shufflevector(lo, hi, 0,1,2,3,4,5,6,7,8,9,10,11,12,13,14,15);
}

// B 32x16 (KxN), stored K-major per column: lane<16 -> col=lane, K = {0..7, 16..23}
//                                           lane>=16 -> col=lane-16, K = {8..15, 24..31}
__device__ __forceinline__ v16bf load_B_frag(const bf16* __restrict__ base, int ld, int lane) {
  const int half = lane >> 4;
  const int c    = lane & 15;
  const bf16* p  = base + (size_t)c * ld + half * 8;
  v8bf lo = *(const v8bf*)(p);
  v8bf hi = *(const v8bf*)(p + 16);
  return __builtin_shufflevector(lo, hi, 0,1,2,3,4,5,6,7,8,9,10,11,12,13,14,15);
}

// ---------------------------------------------------------------------------
// TDM: DMA a 2D tile (tileH rows x tileW elems, 2-byte elems, row stride
// strideElems) from global memory into LDS at byte offset lds_off.
// D# layout per CDNA5 ISA (groups 0/1; groups 2/3 zero:
// tile_dim2/3 = 0 so their fields are unused for a 2D tile).
// ---------------------------------------------------------------------------
__device__ __forceinline__ void tdm_load_tile_2d(unsigned lds_off, const void* gaddr,
                                                 unsigned tileW, unsigned tileH,
                                                 unsigned tensorW, unsigned tensorH,
                                                 unsigned strideElems) {
  const unsigned long long ga = (unsigned long long)gaddr;
  v4u g0;
  g0[0] = 1u;                                                    // count=1, user D#
  g0[1] = lds_off;                                               // lds_addr [63:32]
  g0[2] = (unsigned)(ga & 0xFFFFFFFFu);                          // global_addr lo
  g0[3] = (unsigned)((ga >> 32) & 0x01FFFFFFu) | (2u << 30);     // addr hi | type=2
  v8u g1;
  g1[0] = (1u << 16);                                            // data_size=1 (2B), mask=0
  g1[1] = (tensorW & 0xFFFFu) << 16;                             // tensor_dim0 lo16
  g1[2] = ((tensorW >> 16) & 0xFFFFu) | ((tensorH & 0xFFFFu) << 16); // dim0 hi | dim1 lo
  g1[3] = ((tensorH >> 16) & 0xFFFFu) | (tileW << 16);           // dim1 hi | tile_dim0
  g1[4] = tileH & 0xFFFFu;                                       // tile_dim1, tile_dim2=0
  g1[5] = strideElems;                                           // tensor_dim0_stride lo32
  g1[6] = 0u;                                                    // stride hi | dim1_stride lo
  g1[7] = 0u;
  v4u gz = {0u, 0u, 0u, 0u};
  asm volatile("tensor_load_to_lds %0, %1, %2, %3"
               :: "s"(g0), "s"(g1), "s"(gz), "s"(gz)
               : "memory");
}

// ---------------------------------------------------------------------------
// fp32 -> bf16 convert
// ---------------------------------------------------------------------------
__global__ void f32_to_bf16_kernel(const float* __restrict__ in, bf16* __restrict__ out, int n) {
  int i = blockIdx.x * blockDim.x + threadIdx.x;
  if (i < n) out[i] = (bf16)in[i];
}

// ---------------------------------------------------------------------------
// C[m,n] = sum_k A[m,k] * B[n,k]  (A: MxK bf16, B: NxK bf16, both K-major)
// Block tile 128x128, 8 waves in 4(M)x2(N), wave tile 32x64 (8 WMMAs/K-step).
// A/B 128x32 K-slices are double-buffered in LDS, filled by the TDM:
//   wave0 DMAs the next A tile, wave1 the next B tile (TENSORcnt-tracked),
//   workgroup barrier publishes the buffers.
// BF16OUT selects the epilogue at compile time (no runtime branches).
// ---------------------------------------------------------------------------
template <bool BF16OUT>
__global__ __launch_bounds__(256, 1)
void gemm_bf16_kernel(const bf16* __restrict__ A, const bf16* __restrict__ B,
                      bf16* __restrict__ Cb, float* __restrict__ Cf,
                      const float* __restrict__ bias,
                      int M, int N, int K) {
  __shared__ __align__(128) bf16 As[2][128 * 32];
  __shared__ __align__(128) bf16 Bs[2][128 * 32];

  const int lane = threadIdx.x & 31;
  const int wave = threadIdx.x >> 5;     // 0..7
  const int wm   = wave >> 1;            // 0..3  (M)
  const int wn   = wave & 1;             // 0..1  (N)
  const int mblk = blockIdx.y * 128;
  const int nblk = blockIdx.x * 128;
  const int mw   = wm * 32;
  const int nw   = wn * 64;

  v8f acc[2][4] = {};

  // prologue: DMA tile 0 into buffer 0
  if (wave == 0) {
    tdm_load_tile_2d((unsigned)(size_t)&As[0][0], A + (size_t)mblk * K, 32, 128, K, M, K);
    __builtin_amdgcn_s_wait_tensorcnt(0);
  } else if (wave == 1) {
    tdm_load_tile_2d((unsigned)(size_t)&Bs[0][0], B + (size_t)nblk * K, 32, 128, K, N, K);
    __builtin_amdgcn_s_wait_tensorcnt(0);
  }
  __syncthreads();

  for (int kc = 0; kc < K; kc += 32) {
    const int buf = (kc >> 5) & 1;
    // kick off the next tiles into the other buffer (safe: last read of that
    // buffer was in the previous iteration, published by the trailing barrier)
    if (kc + 32 < K) {
      if (wave == 0)
        tdm_load_tile_2d((unsigned)(size_t)&As[buf ^ 1][0],
                         A + (size_t)mblk * K + (kc + 32), 32, 128, K, M, K);
      else if (wave == 1)
        tdm_load_tile_2d((unsigned)(size_t)&Bs[buf ^ 1][0],
                         B + (size_t)nblk * K + (kc + 32), 32, 128, K, N, K);
    }

    // compute on the current buffer (ds_load fragments, K-major rows of 32)
    v16bf a0 = load_A_frag(&As[buf][(mw     ) * 32], 32, lane);
    v16bf a1 = load_A_frag(&As[buf][(mw + 16) * 32], 32, lane);
    #pragma unroll
    for (int j = 0; j < 4; ++j) {
      v16bf b = load_B_frag(&Bs[buf][(nw + j * 16) * 32], 32, lane);
      acc[0][j] = WMMA_BF16(a0, b, acc[0][j]);
      acc[1][j] = WMMA_BF16(a1, b, acc[1][j]);
    }

    if (kc + 32 < K && wave < 2) __builtin_amdgcn_s_wait_tensorcnt(0);
    __syncthreads();
  }

  // epilogue: C-layout row = v + 8*half, col = lane&15; rows step by N elems
  const int half = lane >> 4;
  const int r    = lane & 15;
  #pragma unroll
  for (int i = 0; i < 2; ++i)
    #pragma unroll
    for (int j = 0; j < 4; ++j) {
      const size_t base = (size_t)(mblk + mw + i * 16 + half * 8) * N
                        + (size_t)(nblk + nw + j * 16 + r);
      if (BF16OUT) {
        bf16* p = Cb + base;
        #pragma unroll
        for (int v = 0; v < 8; ++v) p[(size_t)v * N] = (bf16)acc[i][j][v];
      } else {
        const float bv = bias[nblk + nw + j * 16 + r];
        float* p = Cf + base;
        #pragma unroll
        for (int v = 0; v < 8; ++v) p[(size_t)v * N] = acc[i][j][v] + bv;
      }
    }
}

// ---------------------------------------------------------------------------
// RoPE: out[s,h,i]    = e*cos - o*sin   (i in 0..63, e=in[2i], o=in[2i+1])
//       out[s,h,i+64] = o*cos + e*sin
// ---------------------------------------------------------------------------
__global__ void rope_kernel(const bf16* __restrict__ in, bf16* __restrict__ out,
                            const int* __restrict__ pos, int n) {
  int idx = blockIdx.x * blockDim.x + threadIdx.x;
  if (idx >= n) return;
  const int i   = idx & 63;
  const int h   = (idx >> 6) & (H - 1);
  const int s   = idx >> 10;
  const float p = (float)pos[s];
  const float inv_freq = __powf(10000.0f, -(float)(2 * i) / (float)HD);
  float sn, cs;
  __sincosf(p * inv_freq, &sn, &cs);
  const bf16* row = in  + (size_t)s * D + h * HD;
  bf16*      orow = out + (size_t)s * D + h * HD;
  const float e = (float)row[2 * i];
  const float o = (float)row[2 * i + 1];
  orow[i]      = (bf16)(e * cs - o * sn);
  orow[i + 64] = (bf16)(o * cs + e * sn);
}

// vt[(h*HD+d)*S + s] = v[s*D + h*HD + d]  (V transposed so PV B-frags are contiguous)
__global__ void transpose_v_kernel(const bf16* __restrict__ v, bf16* __restrict__ vt, int n) {
  int idx = blockIdx.x * blockDim.x + threadIdx.x;
  if (idx >= n) return;
  const int s   = idx & (S - 1);
  const int row = idx >> 12;            // S = 4096 = 2^12
  vt[(size_t)row * S + s] = v[(size_t)s * D + row];
}

// ---------------------------------------------------------------------------
// Flash attention, one wave per (head, 16-query tile), sliding-window causal.
// Per-chunk K (32x128) and V^T (128x32) tiles are TDM-DMA'd into LDS with
// per-wave double buffering: issue chunk i+1's two tiles, s_wait_tensorcnt(2)
// (TDM completes in order within a wave), compute chunk i. No barriers needed
// between TDM and compute since one wave owns the whole pipeline.
// __launch_bounds__(32, 1): single wave per block, let regalloc keep the
// 64-VGPR O accumulators + 32-VGPR Q fragments resident (no scratch spills).
// ---------------------------------------------------------------------------
__global__ __launch_bounds__(32, 1)
void flash_attn_kernel(const bf16* __restrict__ Q, const bf16* __restrict__ K,
                       const bf16* __restrict__ Vt, bf16* __restrict__ O) {
  __shared__ __align__(128) bf16 Ks[2][32 * 128];   // keys x head-dim
  __shared__ __align__(128) bf16 Vs[2][128 * 32];   // head-dim x keys
  __shared__ __align__(64)  bf16 pbuf[16 * 32];

  const int lane = threadIdx.x & 31;
  const int head = blockIdx.y;
  const int q0   = blockIdx.x * 16;
  const int half = lane >> 4;
  const int r    = lane & 15;
  const float scale = 0.088388347648318447f;   // 1/sqrt(128)

  v16bf qf[4];
  #pragma unroll
  for (int kk = 0; kk < 4; ++kk)
    qf[kk] = load_A_frag(Q + (size_t)q0 * D + head * HD + kk * 32, D, lane);

  v8f o[8] = {};
  float mst[8], lst[8];
  #pragma unroll
  for (int v = 0; v < 8; ++v) { mst[v] = -1e30f; lst[v] = 0.0f; }

  int kstart = q0 - (WIN - 1);
  if (kstart < 0) kstart = 0;
  kstart &= ~31;

  // prologue: DMA chunk 0 (K tile: 32 key rows x 128 dims; V tile: 128 dim rows x 32 keys)
  tdm_load_tile_2d((unsigned)(size_t)&Ks[0][0], K + (size_t)kstart * D + head * HD,
                   128, 32, D, S, D);
  tdm_load_tile_2d((unsigned)(size_t)&Vs[0][0], Vt + (size_t)(head * HD) * S + kstart,
                   32, 128, S, H * HD, S);

  for (int kc = kstart; kc <= q0 + 15; kc += 32) {
    const int buf = ((kc - kstart) >> 5) & 1;
    if (kc + 32 <= q0 + 15) {   // DMA next chunk, then drain down to the 2 new ops
      tdm_load_tile_2d((unsigned)(size_t)&Ks[buf ^ 1][0],
                       K + (size_t)(kc + 32) * D + head * HD, 128, 32, D, S, D);
      tdm_load_tile_2d((unsigned)(size_t)&Vs[buf ^ 1][0],
                       Vt + (size_t)(head * HD) * S + (kc + 32), 32, 128, S, H * HD, S);
      __builtin_amdgcn_s_wait_tensorcnt(2);
    } else {
      __builtin_amdgcn_s_wait_tensorcnt(0);
    }
    __syncthreads();   // wave-level: order TDM-complete vs ds_loads for the compiler

    // ---- scores: S = Q K^T (two 16-key tiles, fragments from LDS) ----
    v8f sc0 = {}, sc1 = {};
    #pragma unroll
    for (int kk = 0; kk < 4; ++kk) {
      v16bf kf0 = load_B_frag(&Ks[buf][ 0 * 128 + kk * 32], 128, lane);
      v16bf kf1 = load_B_frag(&Ks[buf][16 * 128 + kk * 32], 128, lane);
      sc0 = WMMA_BF16(qf[kk], kf0, sc0);
      sc1 = WMMA_BF16(qf[kk], kf1, sc1);
    }

    // ---- mask + online softmax ----
    #pragma unroll
    for (int v = 0; v < 8; ++v) {
      const int q  = q0 + v + half * 8;
      const int k0 = kc + r;
      const int k1 = kc + 16 + r;
      float e0 = (k0 <= q && (q - k0) < WIN) ? sc0[v] * scale : -1e30f;
      float e1 = (k1 <= q && (q - k1) < WIN) ? sc1[v] * scale : -1e30f;
      float mx = fmaxf(e0, e1);
      #pragma unroll
      for (int off = 8; off >= 1; off >>= 1) mx = fmaxf(mx, __shfl_xor(mx, off, 32));
      const float mnew  = fmaxf(mst[v], mx);
      const float alpha = __expf(mst[v] - mnew);
      const float p0 = __expf(e0 - mnew);
      const float p1 = __expf(e1 - mnew);
      float rs = p0 + p1;
      #pragma unroll
      for (int off = 8; off >= 1; off >>= 1) rs += __shfl_xor(rs, off, 32);
      lst[v] = lst[v] * alpha + rs;
      mst[v] = mnew;
      #pragma unroll
      for (int j = 0; j < 8; ++j) o[j][v] *= alpha;
      pbuf[(v + half * 8) * 32 + r]      = (bf16)p0;
      pbuf[(v + half * 8) * 32 + 16 + r] = (bf16)p1;
    }
    __syncthreads();

    // ---- O += P V  (P via LDS re-layout, V tile rows = head-dims) ----
    v16bf pf = load_A_frag(pbuf, 32, lane);
    #pragma unroll
    for (int j = 0; j < 8; ++j) {
      v16bf vf = load_B_frag(&Vs[buf][(j * 16) * 32], 32, lane);
      o[j] = WMMA_BF16(pf, vf, o[j]);
    }
    __syncthreads();
  }

  #pragma unroll
  for (int v = 0; v < 8; ++v) {
    const float inv_l = 1.0f / lst[v];
    const int   q     = q0 + v + half * 8;
    #pragma unroll
    for (int j = 0; j < 8; ++j)
      O[(size_t)q * D + head * HD + j * 16 + r] = (bf16)(o[j][v] * inv_l);
  }
}

// ---------------------------------------------------------------------------
// launch
// ---------------------------------------------------------------------------
extern "C" void kernel_launch(void* const* d_in, const int* in_sizes, int n_in,
                              void* d_out, int out_size, void* d_ws, size_t ws_size,
                              hipStream_t stream) {
  const float* hs  = (const float*)d_in[0];
  const float* Wq  = (const float*)d_in[1];
  const float* Wk  = (const float*)d_in[2];
  const float* Wv  = (const float*)d_in[3];
  const float* Wc  = (const float*)d_in[4];
  const float* bc  = (const float*)d_in[5];
  const int*   pos = (const int*)d_in[6];
  float* out = (float*)d_out;

  const size_t SD = (size_t)S * D;   // 8M elems
  const size_t DD = (size_t)D * D;   // 4M elems

  char* ws = (char*)d_ws;
  size_t off = 0;
  auto take = [&](size_t elems) { bf16* p = (bf16*)(ws + off); off += elems * sizeof(bf16); return p; };

  bf16* hb  = take(SD);   // hidden bf16
  bf16* wqb = take(DD);
  bf16* wkb = take(DD);
  bf16* wvb = take(DD);
  bf16* wcb = take(DD);
  bf16* qb  = take(SD);   // Q pre-rope; reused as attention output buffer
  bf16* kb  = take(SD);
  bf16* vb  = take(SD);
  bf16* qr  = take(SD);
  bf16* kr  = take(SD);
  bf16* vt  = take(SD);
  bf16* ao  = qb;         // alias: qb dead after RoPE

  // 1) fp32 -> bf16
  f32_to_bf16_kernel<<<(int)((SD + 255) / 256), 256, 0, stream>>>(hs, hb, (int)SD);
  f32_to_bf16_kernel<<<(int)((DD + 255) / 256), 256, 0, stream>>>(Wq, wqb, (int)DD);
  f32_to_bf16_kernel<<<(int)((DD + 255) / 256), 256, 0, stream>>>(Wk, wkb, (int)DD);
  f32_to_bf16_kernel<<<(int)((DD + 255) / 256), 256, 0, stream>>>(Wv, wvb, (int)DD);
  f32_to_bf16_kernel<<<(int)((DD + 255) / 256), 256, 0, stream>>>(Wc, wcb, (int)DD);

  // 2) Q/K/V projections (h @ W.T) on bf16 WMMA + TDM-staged tiles
  dim3 ggrid(D / 128, S / 128);
  gemm_bf16_kernel<true><<<ggrid, 256, 0, stream>>>(hb, wqb, qb, nullptr, nullptr, S, D, D);
  gemm_bf16_kernel<true><<<ggrid, 256, 0, stream>>>(hb, wkb, kb, nullptr, nullptr, S, D, D);
  gemm_bf16_kernel<true><<<ggrid, 256, 0, stream>>>(hb, wvb, vb, nullptr, nullptr, S, D, D);

  // 3) RoPE on Q and K
  const int nrope = S * H * (HD / 2);
  rope_kernel<<<(nrope + 255) / 256, 256, 0, stream>>>(qb, qr, pos, nrope);
  rope_kernel<<<(nrope + 255) / 256, 256, 0, stream>>>(kb, kr, pos, nrope);

  // 4) V transpose for PV B-fragments
  transpose_v_kernel<<<(int)((SD + 255) / 256), 256, 0, stream>>>(vb, vt, (int)SD);

  // 5) flash attention: one wave per (16-query tile, head)
  flash_attn_kernel<<<dim3(S / 16, H), 32, 0, stream>>>(qr, kr, vt, ao);

  // 6) output projection + bias -> fp32 d_out
  gemm_bf16_kernel<false><<<ggrid, 256, 0, stream>>>(ao, wcb, nullptr, out, bc, S, D, D);
}
